// AttnPooling_2138893713786
// MI455X (gfx1250) — compile-verified, matrix operand order checked by reading, not run on hardware
//
#include <hip/hip_runtime.h>
#include <hip/hip_bf16.h>
#include <math.h>

typedef __attribute__((ext_vector_type(16))) __bf16 v16bf;
typedef __attribute__((ext_vector_type(8)))  __bf16 v8bf;
typedef __attribute__((ext_vector_type(8)))  float  v8f;

#define IN_F   128
#define HID_F  128
#define TILE_M 128          // rows of H per block
#define NPB    256          // nodes per block in weighted-sum pass
#define EPSV   1e-16f

// ---------------------------------------------------------------------------
// Hardware tanh (gfx1250 V_TANH_F32). Prefer the clang builtin (hazards are
// handled by the compiler); fall back to inline asm with a v_nop to satisfy
// the TRANS32 "1 independent op before result use" rule.
// ---------------------------------------------------------------------------
__device__ inline float fast_tanh(float x) {
#if __has_builtin(__builtin_amdgcn_tanhf)
    return __builtin_amdgcn_tanhf(x);
#else
    float r;
    asm volatile("v_tanh_f32 %0, %1\n\tv_nop" : "=v"(r) : "v"(x));
    return r;
#endif
}

// ---------------------------------------------------------------------------
// Fragment load from LDS, bf16 16x32 A-layout (ISA 7.12.2):
//   lane 0-15  : row M=lane,    K = {k0+0..7,  k0+16..23}
//   lane 16-31 : row M=lane-16, K = {k0+8..15, k0+24..31}
// B (32x16, K-major per column, i.e. w1 row-major [n][k]) uses the mirrored
// layout, so the same loader works with "row" meaning column N.
// ---------------------------------------------------------------------------
__device__ inline v16bf frag_ld(const __bf16* tile, int lane, int k0) {
    const int r  = lane & 15;
    const int hi = (lane >> 4) << 3;            // 0 or 8
    const __bf16* p = tile + r * 128 + k0 + hi; // 16B aligned
    v8bf lo  = *reinterpret_cast<const v8bf*>(p);
    v8bf hi8 = *reinterpret_cast<const v8bf*>(p + 16);
    v16bf v;
#pragma unroll
    for (int j = 0; j < 8; ++j) { v[j] = lo[j]; v[8 + j] = hi8[j]; }
    return v;
}

__device__ inline __bf16 f2bf(float f) { return (__bf16)f; }

// ---------------------------------------------------------------------------
// K0: init output, segment max (-inf), segment denom (0)
// ---------------------------------------------------------------------------
__global__ void ap_init(float* out, int outN, float* segmax, float* denom, int S) {
    int i = blockIdx.x * blockDim.x + threadIdx.x;
    if (i < outN) out[i] = 0.0f;
    if (i < S) {
        segmax[i] = __int_as_float(0xff800000u);  // -inf
        denom[i]  = 0.0f;
    }
}

// ---------------------------------------------------------------------------
// K1: s[i] = tanh(H[i,:] @ w1^T) @ w2^T   via WMMA bf16
//   block = 256 threads (8 waves), TILE_M=128 rows per block
//   LDS: A tile (H rows, bf16, [m][k]) 32KB + B (w1, bf16, [n][k]) 32KB
// ---------------------------------------------------------------------------
__global__ void __launch_bounds__(256)
ap_score(const float* __restrict__ H, const float* __restrict__ w1,
         const float* __restrict__ w2, float* __restrict__ s, int n) {
    __shared__ __bf16 lA[TILE_M * IN_F];   // 32 KB
    __shared__ __bf16 lB[HID_F * IN_F];    // 32 KB

    const int tid  = threadIdx.x;
    const int lane = tid & 31;
    const int wave = tid >> 5;
    const long long rowBase = (long long)blockIdx.x * TILE_M;

    // stage w1 -> lB (already [n][k] row-major = column-major B, what we need)
#pragma unroll
    for (int idx = tid; idx < (HID_F * IN_F) / 4; idx += 256) {
        float4 f = reinterpret_cast<const float4*>(w1)[idx];
        __bf16* d = &lB[idx * 4];
        d[0] = f2bf(f.x); d[1] = f2bf(f.y); d[2] = f2bf(f.z); d[3] = f2bf(f.w);
    }
    // stage H tile -> lA (zero-pad rows past n)
#pragma unroll
    for (int idx = tid; idx < (TILE_M * IN_F) / 4; idx += 256) {
        int r  = idx >> 5;               // 32 float4 per row
        long long gr = rowBase + r;
        float4 f = make_float4(0.f, 0.f, 0.f, 0.f);
        if (gr < n) f = reinterpret_cast<const float4*>(H)[gr * (IN_F / 4) + (idx & 31)];
        __bf16* d = &lA[idx * 4];
        d[0] = f2bf(f.x); d[1] = f2bf(f.y); d[2] = f2bf(f.z); d[3] = f2bf(f.w);
    }
    __syncthreads();

    // each wave: 16 rows x 128 hidden = 8 accumulator tiles, K in 4 chunks
    v8f acc[8];
#pragma unroll
    for (int t = 0; t < 8; ++t) acc[t] = (v8f){0.f,0.f,0.f,0.f,0.f,0.f,0.f,0.f};

    const __bf16* aBase = &lA[(wave * 16) * IN_F];
#pragma unroll
    for (int kc = 0; kc < 4; ++kc) {
        v16bf a = frag_ld(aBase, lane, kc * 32);
#pragma unroll
        for (int t = 0; t < 8; ++t) {
            v16bf b = frag_ld(&lB[(t * 16) * IN_F], lane, kc * 32);
            acc[t] = __builtin_amdgcn_wmma_f32_16x16x32_bf16(
                false, a, false, b, (short)0, acc[t], false, false);
        }
    }

    // s[row] = sum_n tanh(x[row,n]) * w2[n]
    // C layout: vgpr v, lane<16 -> (M=v,   N=lane), lane>=16 -> (M=v+8, N=lane-16)
    const int col = lane & 15;
    float part[8];
#pragma unroll
    for (int v = 0; v < 8; ++v) part[v] = 0.f;
#pragma unroll
    for (int t = 0; t < 8; ++t) {
        float w2v = w2[t * 16 + col];
#pragma unroll
        for (int v = 0; v < 8; ++v)
            part[v] += fast_tanh(acc[t][v]) * w2v;
    }
    // reduce across the 16 lanes sharing a row (xor within half-wave)
#pragma unroll
    for (int off = 1; off <= 8; off <<= 1) {
#pragma unroll
        for (int v = 0; v < 8; ++v)
            part[v] += __shfl_xor(part[v], off, 32);
    }
    if (col == 0) {
        long long rb = rowBase + wave * 16 + ((lane >> 4) << 3);
#pragma unroll
        for (int v = 0; v < 8; ++v) {
            long long r = rb + v;
            if (r < n) s[r] = part[v];
        }
    }
}

// ---------------------------------------------------------------------------
// K2: per-segment max (monotone int/uint atomic trick, init = -inf)
// ---------------------------------------------------------------------------
__device__ inline void atomicMaxF(float* addr, float val) {
    if (val >= 0.f) atomicMax((int*)addr, __float_as_int(val));
    else            atomicMin((unsigned int*)addr, __float_as_uint(val));
}

__global__ void ap_segmax(const float* __restrict__ s, const int* __restrict__ batch,
                          float* segmax, int n) {
    int i = blockIdx.x * blockDim.x + threadIdx.x;
    if (i < n) atomicMaxF(&segmax[batch[i]], s[i]);
}

// ---------------------------------------------------------------------------
// K3: e = exp(s - segmax);  denom[seg] += e
// ---------------------------------------------------------------------------
__global__ void ap_exp(const float* __restrict__ s, const int* __restrict__ batch,
                       const float* __restrict__ segmax,
                       float* __restrict__ e, float* denom, int n) {
    int i = blockIdx.x * blockDim.x + threadIdx.x;
    if (i < n) {
        int b = batch[i];
        float ev = __expf(s[i] - segmax[b]);
        e[i] = ev;
        unsafeAtomicAdd(&denom[b], ev);
    }
}

// ---------------------------------------------------------------------------
// K5: out[seg,f] += (e[i]/(denom+eps)) * H[i,f]
//   128 threads = 128 features; register accumulation across the sorted run,
//   one atomic flush per (block, segment) boundary.
// ---------------------------------------------------------------------------
__global__ void __launch_bounds__(128)
ap_wsum(const float* __restrict__ H, const int* __restrict__ batch,
        const float* __restrict__ e, const float* __restrict__ denom,
        float* __restrict__ out, int n) {
    const int t = threadIdx.x;                    // feature index
    long long start = (long long)blockIdx.x * NPB;
    long long end   = start + NPB; if (end > n) end = n;
    if (start >= end) return;

    float acc = 0.f;
    int cur = batch[start];
    for (long long i = start; i < end; ++i) {
        int seg = batch[i];                       // uniform across block
        if (seg != cur) {
            unsafeAtomicAdd(&out[(long long)cur * IN_F + t], acc);
            acc = 0.f;
            cur = seg;
        }
        float c = e[i] / (denom[seg] + EPSV);
        acc += c * H[i * IN_F + t];
    }
    unsafeAtomicAdd(&out[(long long)cur * IN_F + t], acc);
}

// ---------------------------------------------------------------------------
extern "C" void kernel_launch(void* const* d_in, const int* in_sizes, int n_in,
                              void* d_out, int out_size, void* d_ws, size_t ws_size,
                              hipStream_t stream) {
    const float* H     = (const float*)d_in[0];
    const int*   batch = (const int*)  d_in[1];
    const float* w1    = (const float*)d_in[2];
    const float* w2    = (const float*)d_in[3];
    float*       out   = (float*)d_out;

    const int n = in_sizes[1];              // N nodes
    const int S = out_size / IN_F;          // num segments

    char*  ws     = (char*)d_ws;
    float* s      = (float*)ws;                                   // N
    float* e      = (float*)(ws + (size_t)n * 4);                 // N
    float* segmax = (float*)(ws + (size_t)n * 8);                 // S
    float* denom  = (float*)(ws + (size_t)n * 8 + (size_t)S * 4); // S

    int initN = out_size > S ? out_size : S;
    ap_init  <<<(initN + 255) / 256, 256, 0, stream>>>(out, out_size, segmax, denom, S);
    ap_score <<<(n + TILE_M - 1) / TILE_M, 256, 0, stream>>>(H, w1, w2, s, n);
    ap_segmax<<<(n + 255) / 256, 256, 0, stream>>>(s, batch, segmax, n);
    ap_exp   <<<(n + 255) / 256, 256, 0, stream>>>(s, batch, segmax, e, denom, n);
    ap_wsum  <<<(n + NPB - 1) / NPB, 128, 0, stream>>>(H, batch, e, denom, out, n);
}
